// LiteratiQuantLinear_61838939127937
// MI455X (gfx1250) — matrix operand order repaired and checked by reading, not compiled, not run on hardware
//
#include <hip/hip_runtime.h>

// ---------------------------------------------------------------------------
// LiteratiQuantLinear on MI455X (gfx1250, wave32, WMMA)
//   out[M,N] = x[M,K] * w_q[N,K]^T,  w_q = sign(w)*max(scale,1e-8) per 128-grp
//   M=8192, N=4096, K=4096. bf16 WMMA (16x16x32), f32 accumulate.
//   Double-buffered LDS, async global->LDS fills (ASYNCcnt) when available.
// ---------------------------------------------------------------------------

typedef __attribute__((ext_vector_type(16))) __bf16 v16bf;
typedef __attribute__((ext_vector_type(8)))  float  v8f;

#define GROUP 128
#define BM 256
#define BN 128
#define BK 32
#define LDP 40   // LDS pitch in bf16 elems (80 B, 16B-aligned, breaks bank conflicts)
#define NFA 4    // A fill chunks (16B) per thread: 256*32/8/256
#define NFB 2    // B fill chunks per thread:       128*32/8/256

#if __has_builtin(__builtin_amdgcn_global_load_async_to_lds_b128)
#define ASYNC_LDS 1
typedef int i32x4 __attribute__((vector_size(16)));
typedef __attribute__((address_space(1))) i32x4* gv4p;   // global int4*
typedef __attribute__((address_space(3))) i32x4* lv4p;   // LDS int4*
#endif

__device__ __forceinline__ void wait_async_lds() {
#if __has_builtin(__builtin_amdgcn_s_wait_asynccnt)
  __builtin_amdgcn_s_wait_asynccnt(0);
#else
  asm volatile("s_wait_asynccnt 0x0" ::: "memory");
#endif
}

__device__ __forceinline__ unsigned short f32_to_bf16(float f) {
  unsigned int u = __float_as_uint(f);
  u += 0x7fffu + ((u >> 16) & 1u);      // round-to-nearest-even
  return (unsigned short)(u >> 16);
}

// ---- prep 1: x f32 -> bf16 (8 elems / thread) -----------------------------
__global__ void cvt_x_bf16(const float* __restrict__ x,
                           unsigned short* __restrict__ xb, int n8) {
  int i = blockIdx.x * blockDim.x + threadIdx.x;
  if (i >= n8) return;
  const float4* p = (const float4*)x + (size_t)i * 2;
  float4 a = p[0], b = p[1];
  uint4 o;
  o.x = (unsigned)f32_to_bf16(a.x) | ((unsigned)f32_to_bf16(a.y) << 16);
  o.y = (unsigned)f32_to_bf16(a.z) | ((unsigned)f32_to_bf16(a.w) << 16);
  o.z = (unsigned)f32_to_bf16(b.x) | ((unsigned)f32_to_bf16(b.y) << 16);
  o.w = (unsigned)f32_to_bf16(b.z) | ((unsigned)f32_to_bf16(b.w) << 16);
  ((uint4*)xb)[i] = o;
}

// ---- prep 2: w_q = sign(w) * clamp(scale) -> bf16 (8 elems / thread) ------
__global__ void quant_w_bf16(const float* __restrict__ w,
                             const float* __restrict__ sc,
                             unsigned short* __restrict__ wq, int n8, int K) {
  int i = blockIdx.x * blockDim.x + threadIdx.x;
  if (i >= n8) return;
  int chunksPerRow = K >> 3;
  int row = i / chunksPerRow;
  int c0  = (i - row * chunksPerRow) << 3;        // 8 elems stay in one group
  float s = sc[(size_t)row * (K / GROUP) + (c0 >> 7)];
  s = fmaxf(s, 1e-8f);
  const float4* p = (const float4*)(w + (size_t)row * K + c0);
  float4 a = p[0], b = p[1];
  uint4 o;
  o.x = (unsigned)f32_to_bf16(a.x >= 0.f ? s : -s) |
        ((unsigned)f32_to_bf16(a.y >= 0.f ? s : -s) << 16);
  o.y = (unsigned)f32_to_bf16(a.z >= 0.f ? s : -s) |
        ((unsigned)f32_to_bf16(a.w >= 0.f ? s : -s) << 16);
  o.z = (unsigned)f32_to_bf16(b.x >= 0.f ? s : -s) |
        ((unsigned)f32_to_bf16(b.y >= 0.f ? s : -s) << 16);
  o.w = (unsigned)f32_to_bf16(b.z >= 0.f ? s : -s) |
        ((unsigned)f32_to_bf16(b.w >= 0.f ? s : -s) << 16);
  ((uint4*)wq)[i] = o;
}

// ---- main GEMM: C[M,N] = A[M,K](bf16) * B[N,K](bf16)^T --------------------
// 256 threads = 8 waves (wave32). Block tile 256x128, waves 4(M) x 2(N),
// each wave: 4x4 tiles of 16x16 -> 16x v_wmma_f32_16x16x32_bf16 per K-step.
// Double-buffered LDS; next tile streamed with async global->LDS.
__global__ __launch_bounds__(256)
void bitnet_gemm_wmma(const unsigned short* __restrict__ A,
                      const unsigned short* __restrict__ B,
                      float* __restrict__ C, int M, int N, int K) {
  __shared__ unsigned short sA[2][BM * LDP];   // 2 x 20 KiB
  __shared__ unsigned short sB[2][BN * LDP];   // 2 x 10 KiB

  const int tid  = threadIdx.x;
  const int lane = tid & 31;
  const int wave = tid >> 5;
  const int m0 = blockIdx.y * BM;
  const int n0 = blockIdx.x * BN;
  const int wm = (wave & 3) * 64;   // wave M offset in block tile
  const int wn = (wave >> 2) * 64;  // wave N offset in block tile

  v8f acc[4][4] = {};

  const int r16   = lane & 15;
  const int hi    = lane >> 4;        // 0 | 1
  const int kselA = hi * 8;           // A: lanes<16 hold K{0..7,16..23}
  const int kselB = hi * 8;           // B: lanes<16 hold K{0..7,16..23}
  union Frag { v16bf v; uint4 q[2]; };

  const unsigned short* gA0 = A + (size_t)m0 * K;
  const unsigned short* gB0 = B + (size_t)n0 * K;

  // cooperative fill of one BK-slice into buffer `buf` (async when available)
  auto fill = [&](int buf, int k0) {
#pragma unroll
    for (int j = 0; j < NFA; ++j) {
      int idx = tid + j * 256;
      int row = idx >> 2, col = (idx & 3) << 3;
      const unsigned short* g = gA0 + (size_t)row * K + k0 + col;
      unsigned short* l = &sA[buf][row * LDP + col];
#if ASYNC_LDS
      __builtin_amdgcn_global_load_async_to_lds_b128((gv4p)g, (lv4p)l, 0, 0);
#else
      *(uint4*)l = *(const uint4*)g;
#endif
    }
#pragma unroll
    for (int j = 0; j < NFB; ++j) {
      int idx = tid + j * 256;
      int row = idx >> 2, col = (idx & 3) << 3;
      const unsigned short* g = gB0 + (size_t)row * K + k0 + col;
      unsigned short* l = &sB[buf][row * LDP + col];
#if ASYNC_LDS
      __builtin_amdgcn_global_load_async_to_lds_b128((gv4p)g, (lv4p)l, 0, 0);
#else
      *(uint4*)l = *(const uint4*)g;
#endif
    }
  };

  const int nIter = K / BK;

  fill(0, 0);
#if ASYNC_LDS
  wait_async_lds();
#endif
  __syncthreads();

  for (int it = 0; it < nIter; ++it) {
    const int cur = it & 1;
    if (it + 1 < nIter) fill(cur ^ 1, (it + 1) * BK);

    const unsigned short* tA = sA[cur];
    const unsigned short* tB = sB[cur];

    Frag fa[4], fb[4];
#pragma unroll
    for (int im = 0; im < 4; ++im) {
      const unsigned short* p = tA + (wm + im * 16 + r16) * LDP + kselA;
      fa[im].q[0] = *(const uint4*)(p);
      fa[im].q[1] = *(const uint4*)(p + 16);
    }
#pragma unroll
    for (int in = 0; in < 4; ++in) {
      const unsigned short* p = tB + (wn + in * 16 + r16) * LDP + kselB;
      fb[in].q[0] = *(const uint4*)(p);
      fb[in].q[1] = *(const uint4*)(p + 16);
    }

#pragma unroll
    for (int im = 0; im < 4; ++im)
#pragma unroll
      for (int in = 0; in < 4; ++in)
        acc[im][in] = __builtin_amdgcn_wmma_f32_16x16x32_bf16(
            false, fa[im].v, false, fb[in].v, (short)0, acc[im][in],
            false, false);

#if ASYNC_LDS
    wait_async_lds();   // fills of next buffer landed in LDS
#endif
    __syncthreads();    // all waves done reading `cur` and done filling `cur^1`
  }

  // epilogue: C/D layout -> lanes<16: M=r, N=lane; lanes>=16: M=r+8, N=lane-16
  const int crow = hi * 8;
#pragma unroll
  for (int im = 0; im < 4; ++im)
#pragma unroll
    for (int in = 0; in < 4; ++in) {
      float* cp = C + (size_t)(m0 + wm + im * 16 + crow) * N
                    + (n0 + wn + in * 16 + r16);
#pragma unroll
      for (int r = 0; r < 8; ++r)
        cp[(size_t)r * N] = acc[im][in][r];
    }
}

// ---------------------------------------------------------------------------
extern "C" void kernel_launch(void* const* d_in, const int* in_sizes, int n_in,
                              void* d_out, int out_size, void* d_ws, size_t ws_size,
                              hipStream_t stream) {
  const float* x  = (const float*)d_in[0];   // (4, 2048, 4096) f32
  const float* w  = (const float*)d_in[1];   // (4096, 4096)    f32
  const float* sc = (const float*)d_in[2];   // (4096, 32)      f32
  float* out = (float*)d_out;                // (4, 2048, 4096) f32

  const int K = 4096;
  const int M = in_sizes[0] / K;             // 8192
  const int N = in_sizes[1] / K;             // 4096

  // scratch: [0, M*K*2) x-bf16 (64 MiB) | [.., +N*K*2) w_q-bf16 (32 MiB)
  unsigned short* xb = (unsigned short*)d_ws;
  unsigned short* wq = xb + (size_t)M * K;

  {
    int n8 = (M * K) / 8;
    cvt_x_bf16<<<(n8 + 255) / 256, 256, 0, stream>>>(x, xb, n8);
  }
  {
    int n8 = (N * K) / 8;
    quant_w_bf16<<<(n8 + 255) / 256, 256, 0, stream>>>(w, sc, wq, n8, K);
  }
  dim3 grid(N / BN, M / BM);
  bitnet_gemm_wmma<<<grid, 256, 0, stream>>>(xb, wq, out, M, N, K);

  (void)n_in; (void)out_size; (void)ws_size;
}
